// QuantumLLM_33603824124335
// MI455X (gfx1250) — matrix operand order repaired and checked by reading, not compiled, not run on hardware
//
#include <hip/hip_runtime.h>
#include <stdint.h>
#include <stddef.h>

// ---------------- problem constants ----------------
constexpr int BATCH = 4;
constexpr int LSEQ  = 4096;
constexpr int DHEAD = 128;
constexpr int BL    = BATCH * LSEQ;        // 16384 rows
constexpr size_t BLD = (size_t)BL * DHEAD; // 2097152 elems per [B,L,D] tensor
constexpr int NQKV  = 6 * DHEAD;           // 768
constexpr int NOUT  = 2 * DHEAD;           // 256
constexpr float SCALE = 0.08838834764831845f; // 128^-0.5

// ---------------- feature probes ----------------
#if defined(__has_builtin)
# if __has_builtin(__builtin_amdgcn_global_load_async_to_lds_b128)
#  define HAS_ASYNC_LDS 1
# endif
# if __has_builtin(__builtin_amdgcn_s_wait_asynccnt)
#  define HAS_WAIT_ASYNC 1
# endif
#endif
#ifndef HAS_ASYNC_LDS
# define HAS_ASYNC_LDS 0
#endif
#ifndef HAS_WAIT_ASYNC
# define HAS_WAIT_ASYNC 0
#endif

// ---------------- WMMA types / helpers ----------------
typedef __attribute__((ext_vector_type(16))) __bf16 v16bf;
typedef __attribute__((ext_vector_type(8)))  float  v8f;
typedef __attribute__((ext_vector_type(4)))  int    v4i;

__device__ __forceinline__ v8f wmma_bf16(v16bf a, v16bf b, v8f c) {
  // D = A(16x32) * B(32x16) + C(16x16), f32 accumulate
  return __builtin_amdgcn_wmma_f32_16x16x32_bf16(
      /*neg_a=*/false, a, /*neg_b=*/false, b,
      /*c_mod=*/(short)0, c, /*reuse_a=*/false, /*reuse_b=*/false);
}

// A-operand (16x32 bf16): lane = 16*half + m, elements: K = 16*half + e
__device__ __forceinline__ v16bf load_A_bf16(const __bf16* row, int half) {
  v16bf a;
  const __bf16* p = row + 16 * half;
#pragma unroll
  for (int e = 0; e < 8; ++e)  a[e] = p[e];
#pragma unroll
  for (int e = 8; e < 16; ++e) a[e] = p[e];
  return a;
}

// Same A layout, fp32 source converted to bf16 on the fly.
__device__ __forceinline__ v16bf load_A_f32(const float* row, int half) {
  v16bf a;
  const float* p = row + 16 * half;
#pragma unroll
  for (int e = 0; e < 8; ++e)  a[e] = (__bf16)p[e];
#pragma unroll
  for (int e = 8; e < 16; ++e) a[e] = (__bf16)p[e];
  return a;
}

// B-operand (32x16 bf16), strided source: lane = 16*khalf + n, elements K strided by ld.
__device__ __forceinline__ v16bf load_B_strided(const __bf16* p, int ld) {
  v16bf b;
#pragma unroll
  for (int e = 0; e < 16; ++e) b[e] = p[(size_t)e * ld];
  return b;
}

// ---------------- async LDS tile staging ----------------
// Copies a 64x128 bf16 tile (16KB) global -> LDS with 128 threads (128B / thread).
__device__ __forceinline__ void tile_copy(__bf16* dst, const __bf16* src, int tid) {
#if HAS_ASYNC_LDS
  typedef __attribute__((address_space(1))) v4i gv4i;
  typedef __attribute__((address_space(3))) v4i lv4i;
#pragma unroll
  for (int i = 0; i < 8; ++i) {
    const int c = (i * 128 + tid) * 8;   // 16-byte chunk index
    __builtin_amdgcn_global_load_async_to_lds_b128(
        (gv4i*)(src + c), (lv4i*)(dst + c), 0, 0);
  }
#else
#pragma unroll
  for (int i = 0; i < 8; ++i) {
    const int c = (i * 128 + tid) * 8;
    *(uint4*)(dst + c) = *(const uint4*)(src + c);
  }
#endif
}

__device__ __forceinline__ void wait_async() {
#if HAS_ASYNC_LDS
# if HAS_WAIT_ASYNC
  __builtin_amdgcn_s_wait_asynccnt(0);
# else
  asm volatile("s_wait_asynccnt 0x0" ::: "memory");
# endif
#endif
}

// ---------------- kernel 0: weight fp32 -> bf16 ----------------
__global__ void cvt_weights(const float* __restrict__ Wq, const float* __restrict__ Wo,
                            __bf16* __restrict__ Wqb, __bf16* __restrict__ Wob) {
  int i = blockIdx.x * blockDim.x + threadIdx.x;
  if (i < NQKV * NQKV) Wqb[i] = (__bf16)Wq[i];
  if (i < NOUT * NOUT) Wob[i] = (__bf16)Wo[i];
}

// ---------------- kernel 1: fused QKV projection ----------------
// Y[16384 x 768] = X @ Wqkv^T + b ; X = concat of the 6 fp32 inputs along features.
__global__ __launch_bounds__(256) void qkv_gemm(
    const float* __restrict__ qr, const float* __restrict__ qi,
    const float* __restrict__ kr, const float* __restrict__ ki,
    const float* __restrict__ vr, const float* __restrict__ vi,
    const float* __restrict__ bias, const __bf16* __restrict__ Wb,
    __bf16* __restrict__ outbase) {
  const float* srcs[6] = {qr, qi, kr, ki, vr, vi};
  const int wave = threadIdx.x >> 5;
  const int lane = threadIdx.x & 31;
  const int half = lane >> 4;
  const int ln   = lane & 15;
  const int m0 = (blockIdx.x * 8 + wave) * 16;
  const int n0 = blockIdx.y * 64;

  v8f acc[4] = {v8f{}, v8f{}, v8f{}, v8f{}};
#pragma unroll
  for (int j = 0; j < 6; ++j) {                 // compile-time source select -> global loads
    const float* arow_base = srcs[j] + (size_t)(m0 + ln) * DHEAD;
#pragma unroll
    for (int cc = 0; cc < 4; ++cc) {
      const int k0 = j * 128 + cc * 32;
      v16bf a = load_A_f32(arow_base + cc * 32, half);
#pragma unroll
      for (int t = 0; t < 4; ++t) {
        // B[k][n] = Wqkv[n][k] -> lane = n, contiguous k (32B-aligned vector load)
        const __bf16* brow = Wb + (size_t)(n0 + t * 16 + ln) * NQKV + k0 + 16 * half;
        v16bf b = *(const v16bf*)brow;
        acc[t] = wmma_bf16(a, b, acc[t]);
      }
    }
  }
#pragma unroll
  for (int t = 0; t < 4; ++t) {
    const int ncol = n0 + t * 16 + ln;
    const int j = ncol >> 7;                 // which of q_r,q_i,k_r,k_i,v_r,v_i
    __bf16* outp = outbase + (size_t)j * BLD;
    const float bb = bias[ncol];
#pragma unroll
    for (int v = 0; v < 8; ++v) {
      const int m = m0 + v + 8 * half;
      outp[(size_t)m * DHEAD + (ncol & 127)] = (__bf16)(acc[t][v] + bb);
    }
  }
}

// ---------------- kernel 2: block-sparse flash attention ----------------
// grid.x = B * 2 * (L/64); block = 128 threads (4 waves, 16 q-rows each).
// K/V 64x128 tiles double-buffered in LDS via async global->LDS copies.
__global__ __launch_bounds__(128) void attn_kernel(
    const __bf16* __restrict__ qkv, const unsigned char* __restrict__ mask,
    __bf16* __restrict__ Orbase) {
  __shared__ alignas(32) __bf16 ldsK[2][64][128];   // 2 x 16KB
  __shared__ alignas(32) __bf16 ldsV[2][64][128];   // 2 x 16KB
  __shared__ alignas(32) __bf16 ldsP[4][16][64];    // per-wave P transpose staging

  const int tid  = threadIdx.x;
  const int wave = tid >> 5;
  const int lane = tid & 31;
  const int half = lane >> 4;
  const int ln   = lane & 15;

  const int qb = blockIdx.x & 63;
  const int c  = (blockIdx.x >> 6) & 1;
  const int b  = blockIdx.x >> 7;

  const __bf16* Q = qkv + (size_t)(0 + c) * BLD + (size_t)b * LSEQ * DHEAD;
  const __bf16* K = qkv + (size_t)(2 + c) * BLD + (size_t)b * LSEQ * DHEAD;
  const __bf16* V = qkv + (size_t)(4 + c) * BLD + (size_t)b * LSEQ * DHEAD;
  __bf16* O = Orbase + (size_t)c * BLD + (size_t)b * LSEQ * DHEAD;

  const unsigned char* maskrow = mask + (size_t)(qb * 64) * LSEQ;
  const int q0 = qb * 64 + wave * 16;

  // preload Q A-fragments (16 rows x 128 d in 4 chunks of 32)
  v16bf qa[4];
  const __bf16* qrow = Q + (size_t)(q0 + ln) * DHEAD;
#pragma unroll
  for (int dc = 0; dc < 4; ++dc) qa[dc] = load_A_bf16(qrow + dc * 32, half);

  v8f oacc[8];
#pragma unroll
  for (int t = 0; t < 8; ++t) oacc[t] = v8f{};
  float rm[8], rl[8];
#pragma unroll
  for (int v = 0; v < 8; ++v) { rm[v] = -1e30f; rl[v] = 0.0f; }

  // first kept block (diagonal always kept, so one exists)
  int kb_cur = -1;
  for (int t = 0; t <= qb; ++t)
    if (maskrow[(size_t)t * 64]) { kb_cur = t; break; }

  int buf = 0;
  tile_copy(&ldsK[buf][0][0], K + (size_t)kb_cur * 64 * DHEAD, tid);
  tile_copy(&ldsV[buf][0][0], V + (size_t)kb_cur * 64 * DHEAD, tid);

  while (kb_cur >= 0) {
    // scan next kept block (uniform across the workgroup)
    int kb_next = -1;
    for (int t = kb_cur + 1; t <= qb; ++t)
      if (maskrow[(size_t)t * 64]) { kb_next = t; break; }

    wait_async();        // this wave's copies into ldsK/ldsV[buf] complete
    __syncthreads();     // all waves' copies visible; prev reads of buf^1 done

    if (kb_next >= 0) {  // overlap next tile DMA with this block's WMMAs
      tile_copy(&ldsK[buf ^ 1][0][0], K + (size_t)kb_next * 64 * DHEAD, tid);
      tile_copy(&ldsV[buf ^ 1][0][0], V + (size_t)kb_next * 64 * DHEAD, tid);
    }

    // ---- S = Q K^T (16 x 64), f32, B-fragments from LDS ----
    v8f s[4];
#pragma unroll
    for (int nt = 0; nt < 4; ++nt) s[nt] = v8f{};
#pragma unroll
    for (int nt = 0; nt < 4; ++nt) {
#pragma unroll
      for (int dc = 0; dc < 4; ++dc) {
        // B[d][kcol] = K[kcol][d] -> lane = kcol, elements contiguous in d
        const __bf16* brow = &ldsK[buf][nt * 16 + ln][dc * 32 + 16 * half];
        v16bf bfrag = *(const v16bf*)brow;
        s[nt] = wmma_bf16(qa[dc], bfrag, s[nt]);
      }
    }

    // ---- online softmax over the 64 keys ----
    float mnew[8];
#pragma unroll
    for (int v = 0; v < 8; ++v) {
      float mx = rm[v];
#pragma unroll
      for (int nt = 0; nt < 4; ++nt) { s[nt][v] *= SCALE; mx = fmaxf(mx, s[nt][v]); }
      mnew[v] = mx;
    }
#pragma unroll
    for (int off = 1; off < 16; off <<= 1)
#pragma unroll
      for (int v = 0; v < 8; ++v)
        mnew[v] = fmaxf(mnew[v], __shfl_xor(mnew[v], off, 32)); // within 16-lane half

    float rs[8];
#pragma unroll
    for (int v = 0; v < 8; ++v) {
      float sum = 0.0f;
#pragma unroll
      for (int nt = 0; nt < 4; ++nt) {
        float p = __expf(s[nt][v] - mnew[v]);
        s[nt][v] = p;
        sum += p;
      }
      rs[v] = sum;
    }
#pragma unroll
    for (int off = 1; off < 16; off <<= 1)
#pragma unroll
      for (int v = 0; v < 8; ++v)
        rs[v] += __shfl_xor(rs[v], off, 32);

#pragma unroll
    for (int v = 0; v < 8; ++v) {
      const float alpha = __expf(rm[v] - mnew[v]);
      rl[v] = rl[v] * alpha + rs[v];
      rm[v] = mnew[v];
#pragma unroll
      for (int t = 0; t < 8; ++t) oacc[t][v] *= alpha;
    }

    // ---- P: C-layout -> A-layout via per-wave LDS tile ----
#pragma unroll
    for (int v = 0; v < 8; ++v)
#pragma unroll
      for (int nt = 0; nt < 4; ++nt)
        ldsP[wave][v + 8 * half][nt * 16 + ln] = (__bf16)s[nt][v];

    v16bf pa0 = load_A_bf16(&ldsP[wave][ln][0],  half);
    v16bf pa1 = load_A_bf16(&ldsP[wave][ln][32], half);

    // ---- O += P @ V (16 x 128), V B-fragments from LDS ----
#pragma unroll
    for (int t = 0; t < 8; ++t) {
      const __bf16* vb0 = &ldsV[buf][16 * half][t * 16 + ln];
      const __bf16* vb1 = &ldsV[buf][32 + 16 * half][t * 16 + ln];
      oacc[t] = wmma_bf16(pa0, load_B_strided(vb0, DHEAD), oacc[t]);
      oacc[t] = wmma_bf16(pa1, load_B_strided(vb1, DHEAD), oacc[t]);
    }

    buf ^= 1;
    kb_cur = kb_next;
  }

  // ---- normalize and store bf16 ----
#pragma unroll
  for (int t = 0; t < 8; ++t)
#pragma unroll
    for (int v = 0; v < 8; ++v) {
      const float val = oacc[t][v] / rl[v];
      O[(size_t)(q0 + v + 8 * half) * DHEAD + t * 16 + ln] = (__bf16)val;
    }
}

// ---------------- kernel 3: output projection ----------------
// out[16384 x 256] = concat(O_r, O_i) @ Wout^T + b_out ; halves split into d_out.
__global__ __launch_bounds__(256) void out_gemm(
    const __bf16* __restrict__ Or, const __bf16* __restrict__ Oi,
    const __bf16* __restrict__ Wb, const float* __restrict__ bias,
    float* __restrict__ dout) {
  const __bf16* srcs[2] = {Or, Oi};
  const int wave = threadIdx.x >> 5;
  const int lane = threadIdx.x & 31;
  const int half = lane >> 4;
  const int ln   = lane & 15;
  const int m0 = (blockIdx.x * 8 + wave) * 16;
  const int n0 = blockIdx.y * 64;

  v8f acc[4] = {v8f{}, v8f{}, v8f{}, v8f{}};
#pragma unroll
  for (int j = 0; j < 2; ++j) {
    const __bf16* arow_base = srcs[j] + (size_t)(m0 + ln) * DHEAD;
#pragma unroll
    for (int cc = 0; cc < 4; ++cc) {
      const int k0 = j * 128 + cc * 32;
      v16bf a = load_A_bf16(arow_base + cc * 32, half);
#pragma unroll
      for (int t = 0; t < 4; ++t) {
        const __bf16* brow = Wb + (size_t)(n0 + t * 16 + ln) * NOUT + k0 + 16 * half;
        v16bf b = *(const v16bf*)brow;
        acc[t] = wmma_bf16(a, b, acc[t]);
      }
    }
  }
#pragma unroll
  for (int t = 0; t < 4; ++t) {
    const int ncol = n0 + t * 16 + ln;
    const int j = ncol >> 7;            // 0 -> first return tensor, 1 -> second
    const float bb = bias[ncol];
#pragma unroll
    for (int v = 0; v < 8; ++v) {
      const int m = m0 + v + 8 * half;
      dout[(size_t)j * BLD + (size_t)m * DHEAD + (ncol & 127)] = acc[t][v] + bb;
    }
  }
}

// ---------------- host launcher ----------------
extern "C" void kernel_launch(void* const* d_in, const int* in_sizes, int n_in,
                              void* d_out, int out_size, void* d_ws, size_t ws_size,
                              hipStream_t stream) {
  const float* q_r = (const float*)d_in[0];
  const float* q_i = (const float*)d_in[1];
  const float* k_r = (const float*)d_in[2];
  const float* k_i = (const float*)d_in[3];
  const float* v_r = (const float*)d_in[4];
  const float* v_i = (const float*)d_in[5];
  const float* Wq  = (const float*)d_in[6];
  const float* bq  = (const float*)d_in[7];
  const float* Wo  = (const float*)d_in[8];
  const float* bo  = (const float*)d_in[9];
  const unsigned char* mask = (const unsigned char*)d_in[10];

  // workspace layout (all 32B aligned)
  char* ws = (char*)d_ws;
  __bf16* Wqb  = (__bf16*)ws;                                    // 768*768 bf16
  __bf16* Wob  = (__bf16*)(ws + (size_t)NQKV * NQKV * 2);        // 256*256 bf16
  __bf16* qkv  = (__bf16*)(ws + (size_t)NQKV * NQKV * 2 + (size_t)NOUT * NOUT * 2);
  __bf16* Orbf = qkv + 6 * BLD;                                  // O_r then O_i

  // 0) weights -> bf16
  cvt_weights<<<(NQKV * NQKV + 255) / 256, 256, 0, stream>>>(Wq, Wo, Wqb, Wob);

  // 1) fused QKV projection
  dim3 g1(BL / 128, NQKV / 64);
  qkv_gemm<<<g1, 256, 0, stream>>>(q_r, q_i, k_r, k_i, v_r, v_i, bq, Wqb, qkv);

  // 2) block-sparse flash attention, both channels
  attn_kernel<<<BATCH * 2 * (LSEQ / 64), 128, 0, stream>>>(qkv, mask, Orbf);

  // 3) output projection + split into the two return tensors
  dim3 g3(BL / 128, NOUT / 64);
  out_gemm<<<g3, 256, 0, stream>>>(Orbf, Orbf + BLD, Wob, bo, (float*)d_out);
}
